// SA_Layer_53180285059438
// MI455X (gfx1250) — matrix operand order repaired and checked by reading, not compiled, not run on hardware
//
#include <hip/hip_runtime.h>
#include <hip/hip_bf16.h>

#define BB 8
#define CC 256
#define C4 64
#define NN 4096
#define BN_EPS 1e-5f

typedef __attribute__((ext_vector_type(16))) __bf16 bf16x16;
typedef __attribute__((ext_vector_type(8)))  float  f32x8;

static __device__ inline bf16x16 load_a16(const __bf16* p0, const __bf16* p1) {
  bf16x16 r;
#pragma unroll
  for (int i = 0; i < 8; ++i) { r[i] = p0[i]; r[i + 8] = p1[i]; }
  return r;
}
static __device__ inline bf16x16 load_b16(const __bf16* p) {
  bf16x16 r;
#pragma unroll
  for (int i = 0; i < 16; ++i) r[i] = p[i];
  return r;
}
static __device__ inline f32x8 wmma_bf16(bf16x16 a, bf16x16 b, f32x8 c) {
  return __builtin_amdgcn_wmma_f32_16x16x32_bf16(false, a, false, b, (short)0, c, false, false);
}

// ---------------- Kernel A: Q/V projections (fp32), emit bf16 Q/V ----------------
__global__ __launch_bounds__(256) void ka_project(const float* __restrict__ x,
                                                  const float* __restrict__ w_qk,
                                                  const float* __restrict__ w_v,
                                                  const float* __restrict__ b_v,
                                                  __bf16* __restrict__ Qbf,
                                                  __bf16* __restrict__ Vbf) {
  __shared__ float xs[CC][64];
  const int b = blockIdx.y;
  const int n0 = blockIdx.x * 64;
  const float* xb = x + (size_t)b * CC * NN;
  for (int idx = threadIdx.x; idx < CC * 64; idx += 256) {
    int c = idx >> 6, nn = idx & 63;
    xs[c][nn] = xb[(size_t)c * NN + n0 + nn];
  }
  __syncthreads();
  const int g = threadIdx.x >> 6, nn = threadIdx.x & 63, n = n0 + nn;
  // Q: 16 outputs per thread-group of 64
  for (int oo = 0; oo < 16; ++oo) {
    int o = g * 16 + oo;
    const float* wr = w_qk + (size_t)o * CC;
    float s = 0.0f;
#pragma unroll 4
    for (int c = 0; c < CC; ++c) s = fmaf(wr[c], xs[c][nn], s);
    Qbf[((size_t)b * NN + n) * C4 + o] = (__bf16)s;
  }
  // V: 64 outputs per thread-group
  for (int oo = 0; oo < 64; ++oo) {
    int o = g * 64 + oo;
    const float* wr = w_v + (size_t)o * CC;
    float s = b_v[o];
#pragma unroll 4
    for (int c = 0; c < CC; ++c) s = fmaf(wr[c], xs[c][nn], s);
    Vbf[((size_t)b * CC + o) * NN + n] = (__bf16)s;
  }
}

// ---------------- Kernel B: softmax row stats via WMMA energy tiles ----------------
__global__ __launch_bounds__(128) void kb_row_stats(const __bf16* __restrict__ Qbf,
                                                    float* __restrict__ rmax,
                                                    float* __restrict__ rrs) {
  const int b = blockIdx.y;
  const int wave = threadIdx.x >> 5;
  const int lane = threadIdx.x & 31;
  const int rblk = blockIdx.x * 64 + wave * 16;
  const int col = lane & 15;
  const int hi = lane >> 4;
  const __bf16* Qb = Qbf + (size_t)b * NN * C4;

  // A operand: rows fixed for this wave (lane = row, kbase = hi*16)
  const __bf16* qr = Qb + (size_t)(rblk + col) * C4;
  bf16x16 a0 = load_a16(qr + hi * 16,      qr + 8 + hi * 16);
  bf16x16 a1 = load_a16(qr + 32 + hi * 16, qr + 40 + hi * 16);

  float mx[8], sm[8];
#pragma unroll
  for (int j = 0; j < 8; ++j) { mx[j] = -3.0e38f; sm[j] = 0.0f; }

  for (int mt = 0; mt < NN; mt += 16) {
    const __bf16* qm = Qb + (size_t)(mt + col) * C4 + hi * 16;
    bf16x16 b0 = load_b16(qm);
    bf16x16 b1 = load_b16(qm + 32);
    f32x8 e = (f32x8){0, 0, 0, 0, 0, 0, 0, 0};
    e = wmma_bf16(a0, b0, e);
    e = wmma_bf16(a1, b1, e);
#pragma unroll
    for (int j = 0; j < 8; ++j) {
      float v = e[j];
      float M = fmaxf(mx[j], v);
      sm[j] = sm[j] * __expf(mx[j] - M) + __expf(v - M);
      mx[j] = M;
    }
  }
  // reduce over the 16 lanes sharing each row (xor masks < 16 stay in-half)
#pragma unroll
  for (int j = 0; j < 8; ++j) {
    float m = mx[j], s = sm[j];
#pragma unroll
    for (int off = 1; off < 16; off <<= 1) {
      float om = __shfl_xor(m, off);
      float os = __shfl_xor(s, off);
      float M = fmaxf(m, om);
      s = s * __expf(m - M) + os * __expf(om - M);
      m = M;
    }
    mx[j] = m; sm[j] = s;
  }
  if (col == 0) {
    int base = rblk + hi * 8;
#pragma unroll
    for (int j = 0; j < 8; ++j) {
      rmax[(size_t)b * NN + base + j] = mx[j];
      rrs[(size_t)b * NN + base + j] = 1.0f / sm[j];
    }
  }
}

// ---------------- Kernel C: x_r = V @ softmax-weights, fused recompute ----------------
__global__ __launch_bounds__(128) void kc_attn_apply(const __bf16* __restrict__ Qbf,
                                                     const __bf16* __restrict__ Vbf,
                                                     const float* __restrict__ rmax,
                                                     const float* __restrict__ rrs,
                                                     float* __restrict__ xr) {
  const int b = blockIdx.y;
  const int wave = threadIdx.x >> 5;
  const int lane = threadIdx.x & 31;
  const int m0 = (blockIdx.x * 4 + wave) * 16;
  const int col = lane & 15;
  const int hi = lane >> 4;

  __shared__ __bf16 wlds[4][16][36];  // [wave][col][k], padded stride vs bank conflicts

  const __bf16* Qb = Qbf + (size_t)b * NN * C4;
  const __bf16* Vb = Vbf + (size_t)b * CC * NN;
  const float* mb = rmax + (size_t)b * NN;
  const float* rb = rrs + (size_t)b * NN;

  // energy B operand = Q rows of our m-tile (fixed per wave)
  const __bf16* qm = Qb + (size_t)(m0 + col) * C4 + hi * 16;
  bf16x16 be0 = load_b16(qm);
  bf16x16 be1 = load_b16(qm + 32);

  f32x8 acc[16];
#pragma unroll
  for (int cb = 0; cb < 16; ++cb) acc[cb] = (f32x8){0, 0, 0, 0, 0, 0, 0, 0};

  for (int nc = 0; nc < NN; nc += 32) {
    // per-row stats of this n-chunk: lane L caches n = nc+L, distribute via shfl
    float mAll = mb[nc + lane];
    float rAll = rb[nc + lane];

    const __bf16* qn0 = Qb + (size_t)(nc + col) * C4;
    const __bf16* qn1 = Qb + (size_t)(nc + 16 + col) * C4;
    __builtin_prefetch(qn0 + 32 * C4, 0, 0);  // next chunk's Q rows
    bf16x16 a00 = load_a16(qn0 + hi * 16,      qn0 + 8 + hi * 16);
    bf16x16 a01 = load_a16(qn0 + 32 + hi * 16, qn0 + 40 + hi * 16);
    bf16x16 a10 = load_a16(qn1 + hi * 16,      qn1 + 8 + hi * 16);
    bf16x16 a11 = load_a16(qn1 + 32 + hi * 16, qn1 + 40 + hi * 16);

    f32x8 e0 = (f32x8){0, 0, 0, 0, 0, 0, 0, 0};
    e0 = wmma_bf16(a00, be0, e0);
    e0 = wmma_bf16(a01, be1, e0);
    f32x8 e1 = (f32x8){0, 0, 0, 0, 0, 0, 0, 0};
    e1 = wmma_bf16(a10, be0, e1);
    e1 = wmma_bf16(a11, be1, e1);

#pragma unroll
    for (int j = 0; j < 8; ++j) {
      int r0 = hi * 8 + j;                 // row within chunk, tile0
      float ms0 = __shfl(mAll, r0);
      float rs0 = __shfl(rAll, r0);
      wlds[wave][col][r0] = (__bf16)(__expf(e0[j] - ms0) * rs0);
      float ms1 = __shfl(mAll, 16 + r0);   // tile1 rows
      float rs1 = __shfl(rAll, 16 + r0);
      wlds[wave][col][16 + r0] = (__bf16)(__expf(e1[j] - ms1) * rs1);
    }
    // same-wave LDS is in-order: safe to read back without a barrier
    bf16x16 b2 = load_b16(&wlds[wave][col][hi * 16]);

#pragma unroll
    for (int cb = 0; cb < 16; ++cb) {
      const __bf16* vp = Vb + (size_t)(cb * 16 + col) * NN + nc;
      bf16x16 a2 = load_a16(vp + hi * 16, vp + 8 + hi * 16);
      acc[cb] = wmma_bf16(a2, b2, acc[cb]);
    }
  }

  float* xrb = xr + (size_t)b * CC * NN;
#pragma unroll
  for (int cb = 0; cb < 16; ++cb) {
#pragma unroll
    for (int j = 0; j < 8; ++j) {
      xrb[(size_t)(cb * 16 + hi * 8 + j) * NN + m0 + col] = acc[cb][j];
    }
  }
}

// ---------------- Kernel D: t = w_t(x - x_r) + b_t; BN + ReLU + residual ----------------
__global__ __launch_bounds__(256) void kd_output(const float* __restrict__ x,
                                                 const float* __restrict__ xr,
                                                 const float* __restrict__ w_t,
                                                 const float* __restrict__ b_t,
                                                 const float* __restrict__ gamma,
                                                 const float* __restrict__ beta,
                                                 const float* __restrict__ rmean,
                                                 const float* __restrict__ rvar,
                                                 float* __restrict__ out) {
  __shared__ float ds[CC][64];
  const int b = blockIdx.y;
  const int n0 = blockIdx.x * 64;
  const float* xb = x + (size_t)b * CC * NN;
  const float* xrb = xr + (size_t)b * CC * NN;
  float* ob = out + (size_t)b * CC * NN;
  for (int idx = threadIdx.x; idx < CC * 64; idx += 256) {
    int c = idx >> 6, nn = idx & 63;
    size_t off = (size_t)c * NN + n0 + nn;
    ds[c][nn] = xb[off] - xrb[off];
  }
  __syncthreads();
  const int g = threadIdx.x >> 6, nn = threadIdx.x & 63, n = n0 + nn;
  for (int oo = 0; oo < 64; ++oo) {
    int o = g * 64 + oo;
    const float* wr = w_t + (size_t)o * CC;
    float s = b_t[o];
#pragma unroll 4
    for (int c = 0; c < CC; ++c) s = fmaf(wr[c], ds[c][nn], s);
    float inv = gamma[o] * rsqrtf(rvar[o] + BN_EPS);
    float bn = s * inv + (beta[o] - rmean[o] * inv);
    size_t off = (size_t)o * NN + n;
    ob[off] = xb[off] + fmaxf(bn, 0.0f);
  }
}

extern "C" void kernel_launch(void* const* d_in, const int* in_sizes, int n_in,
                              void* d_out, int out_size, void* d_ws, size_t ws_size,
                              hipStream_t stream) {
  const float* x     = (const float*)d_in[0];
  const float* w_qk  = (const float*)d_in[1];
  const float* w_v   = (const float*)d_in[2];
  const float* b_v   = (const float*)d_in[3];
  const float* w_t   = (const float*)d_in[4];
  const float* b_t   = (const float*)d_in[5];
  const float* gamma = (const float*)d_in[6];
  const float* beta  = (const float*)d_in[7];
  const float* rmean = (const float*)d_in[8];
  const float* rvar  = (const float*)d_in[9];
  float* out = (float*)d_out;

  char* ws = (char*)d_ws;
  __bf16* Qbf = (__bf16*)ws;                                             // 4 MB
  __bf16* Vbf = (__bf16*)(ws + (size_t)BB * NN * C4 * 2);                // 16 MB
  float*  rmax = (float*)(ws + (size_t)BB * NN * C4 * 2
                             + (size_t)BB * CC * NN * 2);                // 128 KB
  float*  rrs = rmax + (size_t)BB * NN;                                  // 128 KB
  float*  xrb = rrs + (size_t)BB * NN;                                   // 32 MB

  dim3 grid(NN / 64, BB);
  ka_project   <<<grid, 256, 0, stream>>>(x, w_qk, w_v, b_v, Qbf, Vbf);
  kb_row_stats <<<grid, 128, 0, stream>>>(Qbf, rmax, rrs);
  kc_attn_apply<<<grid, 128, 0, stream>>>(Qbf, Vbf, rmax, rrs, xrb);
  kd_output    <<<grid, 256, 0, stream>>>(x, xrb, w_t, b_t, gamma, beta, rmean, rvar, out);
}
